// GCN_58669253263915
// MI455X (gfx1250) — matrix operand order repaired and checked by reading, not compiled, not run on hardware
//
#include <hip/hip_runtime.h>

// ---------------------------------------------------------------------------
// GATv2 (3 layers) + global mean pool for MI455X (gfx1250, wave32, WMMA).
// Dense projections: v_wmma_f32_16x16x32_f16 with f32 accumulate, weights
// pre-transposed/padded/f16-converted so both fragments load as b128.
// Edge phase (dominant, ~2.5 GB/layer): vectorized gathers + f32 atomics.
// NOTE: NN is an exact multiple of 16 -> no row masking anywhere in the GEMM.
// ---------------------------------------------------------------------------

#define NN      50000      // nodes (== 16 * 3125, exact tile multiple)
#define NE      1600000    // edges (without self loops)
#define ETOT    (NE + NN)  // edges + self loops
#define NGRAPH  64
#define INDIM   128
#define FOUT    10         // final per-graph output dim

typedef __attribute__((ext_vector_type(16))) _Float16 v16h;
typedef __attribute__((ext_vector_type(8)))  float    v8f;

// ----- ordered-uint encoding for float atomic max (handles negatives) ------
__device__ __forceinline__ unsigned f2ord(float f) {
  unsigned u = __float_as_uint(f);
  return (u & 0x80000000u) ? ~u : (u | 0x80000000u);
}
__device__ __forceinline__ float ord2f(unsigned u) {
  return (u & 0x80000000u) ? __uint_as_float(u & 0x7fffffffu)
                           : __uint_as_float(~u);
}

// ---------------------------------------------------------------------------
// Weight prep: Wt[c * K + k] = f16( c < Dout ? W[k * Dout + c] : 0 )
// for c in [0, Dpad), Dpad = ceil16(Dout). One-shot, tiny.
// ---------------------------------------------------------------------------
__global__ void prep_wt(const float* __restrict__ W, _Float16* __restrict__ Wt,
                        int K, int Dout, int Dpad) {
  int i = blockIdx.x * blockDim.x + threadIdx.x;
  if (i >= Dpad * K) return;
  int c = i / K, k = i - c * K;
  Wt[i] = (_Float16)((c < Dout) ? W[(size_t)k * Dout + c] : 0.0f);
}

// ---------------------------------------------------------------------------
// One K=32 WMMA step. VGPR layouts used here (contiguous k, not the ISA table order):
//   A 16x32 f16 : lane m (0-15) M=m; elems 0..15 -> K = hf*16 + i
//                 (hf = lane>>4), matching B
//   B 32x16 f16 : lane n holds column N = n&15; elems 0..15 -> K = hf*16 + i
//   C/D 16x16 f32: elem j -> row = j + 8*hf, col = lane&15
// ---------------------------------------------------------------------------
__device__ __forceinline__ v8f wmma_step(const float* __restrict__ xrow,
                                         const _Float16* __restrict__ wcol,
                                         int k0, int hf, v8f acc) {
  const float* xp = xrow + k0 + hf * 16;
  float4 a0 = *(const float4*)(xp + 0);
  float4 a1 = *(const float4*)(xp + 4);
  float4 a2 = *(const float4*)(xp + 8);
  float4 a3 = *(const float4*)(xp + 12);
  v16h a;
  a[0]  = (_Float16)a0.x; a[1]  = (_Float16)a0.y;
  a[2]  = (_Float16)a0.z; a[3]  = (_Float16)a0.w;
  a[4]  = (_Float16)a1.x; a[5]  = (_Float16)a1.y;
  a[6]  = (_Float16)a1.z; a[7]  = (_Float16)a1.w;
  a[8]  = (_Float16)a2.x; a[9]  = (_Float16)a2.y;
  a[10] = (_Float16)a2.z; a[11] = (_Float16)a2.w;
  a[12] = (_Float16)a3.x; a[13] = (_Float16)a3.y;
  a[14] = (_Float16)a3.z; a[15] = (_Float16)a3.w;
  v16h b = *(const v16h*)(wcol + k0 + hf * 16);   // 32B aligned contiguous
  return __builtin_amdgcn_wmma_f32_16x16x32_f16(
      /*neg_a=*/false, a, /*neg_b=*/false, b,
      /*c_mod=*/(short)0, acc, /*reuse_a=*/false, /*reuse_b=*/false);
}

// ---------------------------------------------------------------------------
// Y[N x Dout] = f16wmma( X[N x K] @ Wt^T ) + bias. One wave per 16x16 tile,
// 8 waves per block. K is a multiple of 64 -> 2 wmma per loop body.
// N is an exact multiple of 16 -> no row guard. Column guard is lane-uniform
// across the 8 output rows: hoisted, bias loaded once, plain strided stores.
// ---------------------------------------------------------------------------
__global__ void gemm_bias_wmma(const float* __restrict__ X,
                               const _Float16* __restrict__ Wt,
                               const float* __restrict__ bias,
                               float* __restrict__ Y,
                               int K, int Dout, int rowTiles) {
  const int wave = threadIdx.x >> 5;
  const int lane = threadIdx.x & 31;
  const int rowTile = blockIdx.x * (blockDim.x >> 5) + wave;
  if (rowTile >= rowTiles) return;          // wave-uniform exit, EXEC stays full
  const int hf = lane >> 4;
  const int mrow = lane & 15;
  const int row0 = rowTile * 16;
  const int col0 = blockIdx.y * 16;

  const int rowA = row0 + mrow;             // always < NN (exact multiple)
  const int colB = col0 + mrow;             // always < Dpad (zero padded)

  const float*    xrow = X  + (size_t)rowA * K;
  const _Float16* wcol = Wt + (size_t)colB * K;

  v8f acc = {};
  for (int k0 = 0; k0 < K; k0 += 64) {
    acc = wmma_step(xrow, wcol, k0,      hf, acc);
    acc = wmma_step(xrow, wcol, k0 + 32, hf, acc);
  }

  if (colB < Dout) {                        // all-true when Dout % 16 == 0
    const float bv = bias[colB];
    float* yp = Y + (size_t)(row0 + hf * 8) * Dout + colB;
#pragma unroll
    for (int j = 0; j < 8; ++j) {
      *yp = acc[j] + bv;
      yp += Dout;
    }
  }
}

// ---------------------------------------------------------------------------
// per-layer init: m = ordered(-inf)=0, den = 0, agg = 0
// ---------------------------------------------------------------------------
__global__ void init_layer(unsigned* __restrict__ mbuf, float* __restrict__ den,
                           float* __restrict__ agg, int NH, int NHC) {
  int i = blockIdx.x * blockDim.x + threadIdx.x;
  if (i < NH) { mbuf[i] = 0u; den[i] = 0.0f; }
  if (i < NHC) agg[i] = 0.0f;
}

// ---------------------------------------------------------------------------
// logits: e[eid,h] = att[h,:] . leaky_relu(xl[src,h,:] + xr[dst,h,:], 0.2)
// plus ordered atomic max over dst segments. Templated on H,C so the
// feature-row gathers fully unroll into b128 clauses.
// ---------------------------------------------------------------------------
template <int H, int C>
__global__ void edge_logits(const float* __restrict__ xl,
                            const float* __restrict__ xr,
                            const float* __restrict__ att,
                            const long long* __restrict__ ei,
                            float* __restrict__ ebuf,
                            unsigned* __restrict__ mbuf) {
  int tid = blockIdx.x * blockDim.x + threadIdx.x;
  if (tid >= ETOT * H) return;
  int e = tid / H, h = tid - e * H;
  int src, dst;
  if (e < NE) { src = (int)ei[e]; dst = (int)ei[NE + e]; }
  else        { src = dst = e - NE; }
  const float* pl = xl + (size_t)src * (H * C) + h * C;
  const float* pr = xr + (size_t)dst * (H * C) + h * C;
  const float* pa = att + h * C;
  float acc = 0.0f;
#pragma unroll
  for (int c = 0; c < C; ++c) {
    float v = pl[c] + pr[c];
    v = (v > 0.0f) ? v : 0.2f * v;
    acc += pa[c] * v;
  }
  ebuf[tid] = acc;
  atomicMax(&mbuf[dst * H + h], f2ord(acc));
}

// ---------------------------------------------------------------------------
// p = exp(e - max[dst]); den[dst] += p   (p overwrites e in place)
// ---------------------------------------------------------------------------
template <int H>
__global__ void edge_softmax_num(const long long* __restrict__ ei,
                                 float* __restrict__ ebuf,
                                 const unsigned* __restrict__ mbuf,
                                 float* __restrict__ den) {
  int tid = blockIdx.x * blockDim.x + threadIdx.x;
  if (tid >= ETOT * H) return;
  int e = tid / H, h = tid - e * H;
  int dst = (e < NE) ? (int)ei[NE + e] : (e - NE);
  float p = __expf(ebuf[tid] - ord2f(mbuf[dst * H + h]));
  ebuf[tid] = p;
  atomicAdd(&den[dst * H + h], p);
}

// ---------------------------------------------------------------------------
// agg[dst,h,:] += (p / den[dst,h]) * xl[src,h,:]
// ---------------------------------------------------------------------------
template <int H, int C>
__global__ void edge_aggregate(const float* __restrict__ xl,
                               const long long* __restrict__ ei,
                               const float* __restrict__ ebuf,
                               const float* __restrict__ den,
                               float* __restrict__ agg) {
  int tid = blockIdx.x * blockDim.x + threadIdx.x;
  if (tid >= ETOT * H) return;
  int e = tid / H, h = tid - e * H;
  int src, dst;
  if (e < NE) { src = (int)ei[e]; dst = (int)ei[NE + e]; }
  else        { src = dst = e - NE; }
  float alpha = ebuf[tid] / den[dst * H + h];
  const float* pl = xl  + (size_t)src * (H * C) + h * C;
  float*       po = agg + (size_t)dst * (H * C) + h * C;
#pragma unroll
  for (int c = 0; c < C; ++c) atomicAdd(&po[c], alpha * pl[c]);
}

// ---------------------------------------------------------------------------
// buf += bias, optional leaky_relu(0.01)
// ---------------------------------------------------------------------------
__global__ void bias_act(float* __restrict__ buf, const float* __restrict__ b,
                         int cols, int act) {
  int i = blockIdx.x * blockDim.x + threadIdx.x;
  if (i >= NN * cols) return;
  float v = buf[i] + b[i % cols];
  if (act) v = (v > 0.0f) ? v : 0.01f * v;
  buf[i] = v;
}

// ---------------------------------------------------------------------------
// global mean pool
// ---------------------------------------------------------------------------
__global__ void pool_zero(float* __restrict__ sums, float* __restrict__ cnts) {
  int i = blockIdx.x * blockDim.x + threadIdx.x;
  if (i < NGRAPH * FOUT) sums[i] = 0.0f;
  if (i < NGRAPH) cnts[i] = 0.0f;
}
__global__ void pool_acc(const float* __restrict__ h,
                         const long long* __restrict__ batch,
                         float* __restrict__ sums, float* __restrict__ cnts) {
  int n = blockIdx.x * blockDim.x + threadIdx.x;
  if (n >= NN) return;
  int g = (int)batch[n];
#pragma unroll
  for (int j = 0; j < FOUT; ++j)
    atomicAdd(&sums[g * FOUT + j], h[(size_t)n * FOUT + j]);
  atomicAdd(&cnts[g], 1.0f);
}
__global__ void pool_final(const float* __restrict__ sums,
                           const float* __restrict__ cnts,
                           float* __restrict__ out) {
  int i = blockIdx.x * blockDim.x + threadIdx.x;
  if (i >= NGRAPH * FOUT) return;
  out[i] = sums[i] / fmaxf(cnts[i / FOUT], 1.0f);
}

// ---------------------------------------------------------------------------
extern "C" void kernel_launch(void* const* d_in, const int* in_sizes, int n_in,
                              void* d_out, int out_size, void* d_ws, size_t ws_size,
                              hipStream_t stream) {
  const float*     x     = (const float*)d_in[0];
  const long long* ei    = (const long long*)d_in[1];
  const long long* batch = (const long long*)d_in[2];
  // per-layer params start at 3: Wl, bl, Wr, br, att, b
  const float* P[18];
  for (int i = 0; i < 18; ++i) P[i] = (const float*)d_in[3 + i];

  // ---- workspace carve-up (floats; all offsets keep >=32B alignment) ----
  float* f = (float*)d_ws;
  size_t o = 0;
  float*    xl   = f + o; o += (size_t)NN * 64;
  float*    xr   = f + o; o += (size_t)NN * 64;
  float*    bufA = f + o; o += (size_t)NN * 64;   // layer0 out / layer1 in
  float*    bufB = f + o; o += (size_t)NN * 64;   // layer1 out / layer2 in
  float*    bufC = f + o; o += (size_t)NN * 16;   // layer2 out [NN x 10]
  float*    ebuf = f + o; o += (size_t)ETOT * 2;  // logits -> probs, [ETOT*H]
  unsigned* mbuf = (unsigned*)(f + o); o += (size_t)NN * 2;
  float*    den  = f + o; o += (size_t)NN * 2;
  float*    sums = f + o; o += NGRAPH * FOUT;
  float*    cnts = f + o; o += NGRAPH + 8;        // pad to keep alignment
  _Float16* wtL  = (_Float16*)(f + o); o += (128 * 128) / 2;  // f16 halves
  _Float16* wtR  = (_Float16*)(f + o); o += (128 * 128) / 2;

  const int rowTiles = NN / 16;                   // exact

  auto run_gemms = [&](const float* xin, int K, const float* Wl, const float* bl,
                       const float* Wr, const float* br, int Dout) {
    int Dpad = ((Dout + 15) / 16) * 16;
    int nw = Dpad * K;
    prep_wt<<<(nw + 255) / 256, 256, 0, stream>>>(Wl, wtL, K, Dout, Dpad);
    prep_wt<<<(nw + 255) / 256, 256, 0, stream>>>(Wr, wtR, K, Dout, Dpad);
    dim3 gg((rowTiles + 7) / 8, Dpad / 16);
    gemm_bias_wmma<<<gg, 256, 0, stream>>>(xin, wtL, bl, xl, K, Dout, rowTiles);
    gemm_bias_wmma<<<gg, 256, 0, stream>>>(xin, wtR, br, xr, K, Dout, rowTiles);
  };

  auto run_edges = [&](const float* att, const float* b, float* agg,
                       int H, int C, auto logitsK, auto softK, auto aggK, int act) {
    int Dout = H * C;
    int NH = NN * H, NHC = NN * Dout;
    int mx = (NH > NHC) ? NH : NHC;
    init_layer<<<(mx + 255) / 256, 256, 0, stream>>>(mbuf, den, agg, NH, NHC);
    int tE = ETOT * H;
    logitsK<<<(tE + 255) / 256, 256, 0, stream>>>(xl, xr, att, ei, ebuf, mbuf);
    softK<<<(tE + 255) / 256, 256, 0, stream>>>(ei, ebuf, mbuf, den);
    aggK<<<(tE + 255) / 256, 256, 0, stream>>>(xl, ei, ebuf, den, agg);
    bias_act<<<(NHC + 255) / 256, 256, 0, stream>>>(agg, b, Dout, act);
  };

  // ---- layer 0: 128 -> (2 heads x 32), concat, leaky_relu(0.01) ----
  run_gemms(x, INDIM, P[0], P[1], P[2], P[3], 64);
  run_edges(P[4], P[5], bufA, 2, 32,
            edge_logits<2, 32>, edge_softmax_num<2>, edge_aggregate<2, 32>, 1);

  // ---- layer 1: 64 -> (2 x 32), concat, leaky_relu(0.01) ----
  run_gemms(bufA, 64, P[6], P[7], P[8], P[9], 64);
  run_edges(P[10], P[11], bufB, 2, 32,
            edge_logits<2, 32>, edge_softmax_num<2>, edge_aggregate<2, 32>, 1);

  // ---- layer 2: 64 -> (1 x 10), mean over 1 head == identity, no act ----
  run_gemms(bufB, 64, P[12], P[13], P[14], P[15], 10);
  run_edges(P[16], P[17], bufC, 1, 10,
            edge_logits<1, 10>, edge_softmax_num<1>, edge_aggregate<1, 10>, 0);

  // ---- global mean pool ----
  pool_zero<<<(NGRAPH * FOUT + 255) / 256, 256, 0, stream>>>(sums, cnts);
  pool_acc<<<(NN + 255) / 256, 256, 0, stream>>>(bufC, batch, sums, cnts);
  pool_final<<<(NGRAPH * FOUT + 255) / 256, 256, 0, stream>>>(sums, cnts, (float*)d_out);
}
